// ClassicalSelfAttention_65481071406921
// MI455X (gfx1250) — matrix operand order repaired and checked by reading, not compiled, not run on hardware
//
#include <hip/hip_runtime.h>

typedef unsigned int u32;
typedef unsigned short u16;
typedef unsigned long long u64;
typedef __attribute__((ext_vector_type(16))) __bf16 v16bf;
typedef __attribute__((ext_vector_type(8)))  float  v8f;
typedef __attribute__((ext_vector_type(4)))  unsigned int v4u;
typedef __attribute__((ext_vector_type(8)))  int v8i;
typedef __attribute__((ext_vector_type(4)))  int v4i;

#define EMBED 1024
#define NTOK  8192
#define LDS_STRIDE 40   // 32 data + 8 pad ushorts; 80B pitch = 20 DWORDs (16B aligned)

// ---------- bf16 helpers (round-to-nearest-even) ----------
__device__ __forceinline__ u16 f2b(float f) {
  u32 u = __float_as_uint(f);
  u32 r = u + 0x7fffu + ((u >> 16) & 1u);
  return (u16)(r >> 16);
}
__device__ __forceinline__ float b2f(u32 h) { return __uint_as_float(h << 16); }

union Frag { u32 u[8]; v16bf v; };

// ---------- gfx1250 async global->LDS copy (ASYNCcnt-tracked) ----------
__device__ __forceinline__ void async_g2l_b128(u16* lds_dst, const u16* gsrc) {
  const u32 loff = (u32)(size_t)(void*)lds_dst;   // low 32 bits of generic ptr = LDS offset
  const u64 ga = (u64)(size_t)gsrc;
  asm volatile("global_load_async_to_lds_b128 %0, %1, off"
               :: "v"(loff), "v"(ga)
               : "memory");
}
__device__ __forceinline__ void wait_async0() {
  asm volatile("s_wait_asynccnt 0x0" ::: "memory");
}

// ---------- gfx1250 Tensor Data Mover: 128-row x 32-elem bf16 tile -> LDS ----------
// Hardware pad insertion reproduces the LDS pitch: 16 DWORDs data + 4 DWORDs pad = 40 ushorts.
__device__ __forceinline__ void tdm_load_tile_b16(u16* lds_dst, const u16* gsrc,
                                                  u32 ld, u32 rows_total) {
  const u32 lds_off = (u32)(size_t)(void*)lds_dst;
  const u64 ga = (u64)(size_t)gsrc;
  v4u g0;
  g0[0] = 1u;                                   // count=1 (valid user descriptor)
  g0[1] = lds_off;                              // lds_addr (bytes)
  g0[2] = (u32)ga;                              // global_addr[31:0]
  g0[3] = (u32)(ga >> 32) | (2u << 30);         // global_addr[56:32] | type=2 ("image")
  v8i g1;
  g1[0] = (int)((1u << 16)                      // data_size = 2 bytes
              | (1u << 20)                      // pad_enable
              | (3u << 22)                      // pad_interval: 16 DWORDs
              | (3u << 25));                    // pad_amount: 4 DWORDs
  g1[1] = (int)((ld & 0xffffu) << 16);          // tensor_dim0[15:0]   (bits 63:48)
  g1[2] = (int)(((ld >> 16) & 0xffffu)          // tensor_dim0[31:16]  (bits 79:64)
              | ((rows_total & 0xffffu) << 16));// tensor_dim1[15:0]   (bits 95:80)
  g1[3] = (int)(((rows_total >> 16) & 0xffffu)  // tensor_dim1[31:16]  (bits 111:96)
              | (32u << 16));                   // tile_dim0 = 32      (bits 127:112)
  g1[4] = (int)128u;                            // tile_dim1 = 128, tile_dim2 = 0
  g1[5] = (int)ld;                              // tensor_dim0_stride[31:0]
  g1[6] = 0;                                    // stride0[47:32], dim1_stride[15:0]
  g1[7] = 0;                                    // dim1_stride[47:16]
  const v4i z4 = {0, 0, 0, 0};                  // groups 2/3 unused (2D tensor)
  const v8i z8 = {0, 0, 0, 0, 0, 0, 0, 0};      // extra group (clang-23 6-arg form)
  __builtin_amdgcn_tensor_load_to_lds(g0, g1, z4, z4, z8, 0);
}

// ---------- WMMA fragment fills from LDS ----------
// A 16x32 bf16 (ISA 7.12.2): lane L (g=L/16, m=L%16):
// VGPR j in [0,3]: K = 2j+8g,+1 ; VGPR j in [4,7]: K = 16+2(j-4)+8g,+1
__device__ __forceinline__ v16bf load_frag_A(const u16* As, int m0, int lane) {
  const int g = lane >> 4, m = lane & 15;
  const u16* row = As + (m0 + m) * LDS_STRIDE;
  Frag f;
#pragma unroll
  for (int j = 0; j < 4; ++j) f.u[j]     = *(const u32*)(row + 2 * j + 8 * g);
#pragma unroll
  for (int j = 0; j < 4; ++j) f.u[4 + j] = *(const u32*)(row + 16 + 2 * j + 8 * g);
  return f.v;
}

// B 32x16 bf16: lane L (g=L/16, n=L%16), VGPR j in [0,3]: K = 2j+8g,+1 ; VGPR j in [4,7]: K = 16+2(j-4)+8g,+1.
// Bs stored [n][k] so each dword is contiguous -> compiler merges to ds_load_b128.
__device__ __forceinline__ v16bf load_frag_B(const u16* Bs, int n0, int lane) {
  const int g = lane >> 4, n = lane & 15;
  const u16* row = Bs + (n0 + n) * LDS_STRIDE;
  Frag f;
#pragma unroll
  for (int j = 0; j < 8; ++j) f.u[j] = *(const u32*)(row + 2 * (j & 3) + 16 * (j >> 2) + 8 * g);
  return f.v;
}

// ---------- staging (256 threads, 128x32-element tiles) ----------
__device__ __forceinline__ void stage_f32(u16* dst, const float* src, int ld, int t) {
#pragma unroll
  for (int i = 0; i < 4; ++i) {
    const int lin = (i * 256 + t) * 4;
    const int r = lin >> 5, c = lin & 31;
    const float4 v = *(const float4*)(src + (size_t)r * ld + c);
    u16* d = dst + r * LDS_STRIDE + c;
    *(u32*)(d)     = (u32)f2b(v.x) | ((u32)f2b(v.y) << 16);
    *(u32*)(d + 2) = (u32)f2b(v.z) | ((u32)f2b(v.w) << 16);
  }
}

__device__ __forceinline__ void stage_trans_f32(u16* dst, const float* src, int ld, int t) {
#pragma unroll
  for (int i = 0; i < 4; ++i) {
    const int lin = (i * 256 + t) * 4;
    const int k = lin >> 7, n = lin & 127;
    const float4 v = *(const float4*)(src + (size_t)k * ld + n);
    dst[(n + 0) * LDS_STRIDE + k] = f2b(v.x);
    dst[(n + 1) * LDS_STRIDE + k] = f2b(v.y);
    dst[(n + 2) * LDS_STRIDE + k] = f2b(v.z);
    dst[(n + 3) * LDS_STRIDE + k] = f2b(v.w);
  }
}

__device__ __forceinline__ void stage_b16_async(u16* dst, const u16* src, int ld, int t) {
#pragma unroll
  for (int i = 0; i < 2; ++i) {
    const int lin = (i * 256 + t) * 8;
    const int r = lin >> 5, c = lin & 31;
    async_g2l_b128(dst + r * LDS_STRIDE + c, src + (size_t)r * ld + c);
  }
}

// ---------- per-mode tile staging ----------
// MODE 0: A = x (f32), B = w (f32 [k][n], transpose-stage)      VGPR-mediated
// MODE 1: A = q (bf16, ld 1024) via TDM;  B = kb rows via async
// MODE 2: A = P (bf16, ld 8192) via TDM;  B = xT rows via async
template <int MODE>
__device__ __forceinline__ void stage_tile(u16* As, u16* Bs, const void* A, const void* B,
                                           int m0, int n0, int k0, int t) {
  if constexpr (MODE == 0) {
    stage_f32(As, (const float*)A + (size_t)m0 * EMBED + k0, EMBED, t);
    stage_trans_f32(Bs, (const float*)B + (size_t)k0 * EMBED + n0, EMBED, t);
    __builtin_prefetch((const char*)A + ((size_t)(m0 + (t >> 1)) * EMBED + k0 + 32) * 4, 0, 1);
  } else if constexpr (MODE == 1) {
    if ((t >> 5) == 0)  // one wave issues the TDM descriptor for the A tile
      tdm_load_tile_b16(As, (const u16*)A + (size_t)m0 * EMBED + k0, EMBED, NTOK);
    stage_b16_async(Bs, (const u16*)B + (size_t)n0 * EMBED + k0, EMBED, t);
  } else {
    if ((t >> 5) == 0)
      tdm_load_tile_b16(As, (const u16*)A + (size_t)m0 * NTOK + k0, NTOK, NTOK);
    stage_b16_async(Bs, (const u16*)B + (size_t)n0 * NTOK + k0, NTOK, t);
  }
}

template <int MODE>
__device__ __forceinline__ void stage_wait(int t) {
  if constexpr (MODE != 0) {
    if ((t >> 5) == 0) __builtin_amdgcn_s_wait_tensorcnt(0);  // TDM done (issuing wave)
    wait_async0();                                            // per-lane async done
  }
  // MODE 0: ds_store staging is covered by the barrier's DScnt wait
}

// ---------- tiled WMMA GEMM, double-buffered LDS ----------
// MODE 0: C(bf16)[M,1024] = x * w                (projection)
// MODE 1: C(bf16)[M,8192] = q * k^T, scaled      (scores)
// MODE 2: C(f32) [M,1024] = P * x                (output)
template <int MODE>
__global__ __launch_bounds__(256) void wmma_gemm_kernel(const void* __restrict__ A,
                                                        const void* __restrict__ B,
                                                        void* __restrict__ C,
                                                        int K, float scale) {
  __shared__ __align__(16) u16 As[2][128 * LDS_STRIDE];
  __shared__ __align__(16) u16 Bs[2][128 * LDS_STRIDE];

  const int t = threadIdx.x;
  const int lane = t & 31, wid = t >> 5;
  const int wm = wid & 1, wn = wid >> 1;           // 2x4 wave grid -> 64x32 per wave
  const int m0 = blockIdx.y * 128, n0 = blockIdx.x * 128;
  const int ldC = (MODE == 1) ? NTOK : EMBED;

  v8f acc[4][2];
#pragma unroll
  for (int a = 0; a < 4; ++a)
#pragma unroll
    for (int b = 0; b < 2; ++b)
#pragma unroll
      for (int i = 0; i < 8; ++i) acc[a][b][i] = 0.0f;

  // prologue: fill buffer 0
  stage_tile<MODE>(As[0], Bs[0], A, B, m0, n0, 0, t);
  stage_wait<MODE>(t);
  __syncthreads();

  for (int k0 = 0; k0 < K; k0 += 32) {
    const int cur = (k0 >> 5) & 1;

    if (k0 + 32 < K)  // overlap next-tile DMA with this tile's WMMA
      stage_tile<MODE>(As[cur ^ 1], Bs[cur ^ 1], A, B, m0, n0, k0 + 32, t);

    v16bf af[4];
#pragma unroll
    for (int a = 0; a < 4; ++a) af[a] = load_frag_A(As[cur], wm * 64 + a * 16, lane);
    v16bf bfr[2];
#pragma unroll
    for (int b = 0; b < 2; ++b) bfr[b] = load_frag_B(Bs[cur], wn * 32 + b * 16, lane);

#pragma unroll
    for (int a = 0; a < 4; ++a)
#pragma unroll
      for (int b = 0; b < 2; ++b)
        acc[a][b] = __builtin_amdgcn_wmma_f32_16x16x32_bf16(
            false, af[a], false, bfr[b], (short)0, acc[a][b], false, false);

    stage_wait<MODE>(t);
    __syncthreads();
  }

  // C/D layout: lane (g=lane/16, n=lane%16), VGPR r -> (M = r + 8g, N = n)
  const int g = lane >> 4, nl = lane & 15;
#pragma unroll
  for (int a = 0; a < 4; ++a) {
#pragma unroll
    for (int b = 0; b < 2; ++b) {
      const int mbase = m0 + wm * 64 + a * 16 + g * 8;
      const int nbase = n0 + wn * 32 + b * 16 + nl;
#pragma unroll
      for (int r = 0; r < 8; ++r) {
        const float v = acc[a][b][r] * scale;
        if constexpr (MODE == 2)
          ((float*)C)[(size_t)(mbase + r) * ldC + nbase] = v;
        else
          ((u16*)C)[(size_t)(mbase + r) * ldC + nbase] = f2b(v);
      }
    }
  }
}

// ---------- x (f32 [tok][d]) -> xT (bf16 [d][tok]) one-shot transpose+convert ----------
__global__ __launch_bounds__(256) void transpose_conv_kernel(const float* __restrict__ x,
                                                             u16* __restrict__ xT) {
  __shared__ float tile[64][65];
  const int t = threadIdx.x;
  const int tok0 = blockIdx.x * 64, d0 = blockIdx.y * 64;
#pragma unroll
  for (int i = 0; i < 4; ++i) {
    const int lin = (i * 256 + t) * 4;
    const int r = lin >> 6, c = lin & 63;
    const float4 v = *(const float4*)(x + (size_t)(tok0 + r) * EMBED + d0 + c);
    tile[r][c + 0] = v.x; tile[r][c + 1] = v.y;
    tile[r][c + 2] = v.z; tile[r][c + 3] = v.w;
  }
  __syncthreads();
#pragma unroll
  for (int i = 0; i < 4; ++i) {
    const int lin = (i * 256 + t) * 4;
    const int dr = lin >> 6, tc = lin & 63;
    uint2 o;
    o.x = (u32)f2b(tile[tc + 0][dr]) | ((u32)f2b(tile[tc + 1][dr]) << 16);
    o.y = (u32)f2b(tile[tc + 2][dr]) | ((u32)f2b(tile[tc + 3][dr]) << 16);
    *(uint2*)(xT + (size_t)(d0 + dr) * NTOK + tok0 + tc) = o;
  }
}

// ---------- row softmax over bf16 scores, in place ----------
__global__ __launch_bounds__(256) void softmax_kernel(u16* __restrict__ S) {
  const int t = threadIdx.x;
  u16* row = S + (size_t)blockIdx.x * NTOK;

  float v[32];
#pragma unroll
  for (int i = 0; i < 8; ++i) {
    const int c = (i * 256 + t) * 4;
    const uint2 d = *(const uint2*)(row + c);
    v[i * 4 + 0] = b2f(d.x & 0xffffu);
    v[i * 4 + 1] = b2f(d.x >> 16);
    v[i * 4 + 2] = b2f(d.y & 0xffffu);
    v[i * 4 + 3] = b2f(d.y >> 16);
  }

  float mx = -3.4e38f;
#pragma unroll
  for (int j = 0; j < 32; ++j) mx = fmaxf(mx, v[j]);

  __shared__ float red[256];
  red[t] = mx;
  __syncthreads();
  for (int s = 128; s > 0; s >>= 1) {
    if (t < s) red[t] = fmaxf(red[t], red[t + s]);
    __syncthreads();
  }
  mx = red[0];
  __syncthreads();

  float sum = 0.0f;
#pragma unroll
  for (int j = 0; j < 32; ++j) {
    v[j] = __expf(v[j] - mx);
    sum += v[j];
  }
  red[t] = sum;
  __syncthreads();
  for (int s = 128; s > 0; s >>= 1) {
    if (t < s) red[t] += red[t + s];
    __syncthreads();
  }
  const float inv = 1.0f / red[0];

#pragma unroll
  for (int i = 0; i < 8; ++i) {
    const int c = (i * 256 + t) * 4;
    uint2 d;
    d.x = (u32)f2b(v[i * 4 + 0] * inv) | ((u32)f2b(v[i * 4 + 1] * inv) << 16);
    d.y = (u32)f2b(v[i * 4 + 2] * inv) | ((u32)f2b(v[i * 4 + 3] * inv) << 16);
    *(uint2*)(row + c) = d;
  }
}

extern "C" void kernel_launch(void* const* d_in, const int* in_sizes, int n_in,
                              void* d_out, int out_size, void* d_ws, size_t ws_size,
                              hipStream_t stream) {
  (void)in_sizes; (void)n_in; (void)out_size; (void)ws_size;
  const float* x  = (const float*)d_in[0];
  const float* wq = (const float*)d_in[1];
  const float* wk = (const float*)d_in[2];
  float* out = (float*)d_out;

  // Workspace: qb (16 MB) | kb (16 MB) | S/P bf16 (128 MB) | xT bf16 (16 MB)
  u16* qb = (u16*)d_ws;
  u16* kb = qb + (size_t)NTOK * EMBED;
  u16* Sb = kb + (size_t)NTOK * EMBED;
  u16* xT = Sb + (size_t)NTOK * NTOK;

  const dim3 blk(256);
  const float scale = 0.03125f;  // 1/sqrt(1024)

  transpose_conv_kernel<<<dim3(NTOK / 64, EMBED / 64), blk, 0, stream>>>(x, xT);
  wmma_gemm_kernel<0><<<dim3(EMBED / 128, NTOK / 128), blk, 0, stream>>>(x, wq, qb, EMBED, 1.0f);
  wmma_gemm_kernel<0><<<dim3(EMBED / 128, NTOK / 128), blk, 0, stream>>>(x, wk, kb, EMBED, 1.0f);
  wmma_gemm_kernel<1><<<dim3(NTOK / 128, NTOK / 128), blk, 0, stream>>>(qb, kb, Sb, EMBED, scale);
  softmax_kernel<<<NTOK, blk, 0, stream>>>(Sb);
  wmma_gemm_kernel<2><<<dim3(EMBED / 128, NTOK / 128), blk, 0, stream>>>(Sb, xT, out, NTOK, 1.0f);
}
